// MyModel_87522843561131
// MI455X (gfx1250) — matrix operand order repaired and checked by reading, not compiled, not run on hardware
//
#include <hip/hip_runtime.h>
#include <hip/hip_bf16.h>

typedef __attribute__((ext_vector_type(16))) _Float16 v16h;
typedef __attribute__((ext_vector_type(8)))  float    v8f;

#define T_   900
#define H_   400
#define C_   73
#define F_   240
#define B_   32
#define G4H  1600          // 4*H
#define KX   8             // x k-tiles of 32 (240 padded to 256)
#define KH   13            // h k-tiles of 32 (400 padded to 416)
#define KTOT 21            // KX + KH
#define JT   25            // 400/16 j-tiles
#define NTD  5             // dense n-tiles (73 padded to 80)
#define BTC  (B_*T_*C_)    // 2,102,400

// ---- workspace byte offsets ----
#define PW_HALF   (2*JT*4*KTOT*32*16)     // 2,150,400 halves
#define PWD_HALF  (2*NTD*KH*32*16)        //    66,560 halves
#define XA_HALF   (T_*KX*2*32*16)         // 7,372,800 halves
#define OFF_PW    0
#define OFF_PWD   (PW_HALF*2)                     // 4,300,800 B
#define OFF_XA    (OFF_PWD + PWD_HALF*2)          // 4,433,920 B
#define OFF_MAX   (OFF_XA + XA_HALF*2)            // 19,179,520 B

__device__ __forceinline__ float sigmf(float x) {
    return 1.0f / (1.0f + __expf(-x));
}

// A-operand (16-bit, 16x32) lane/half -> k mapping:
// lane L: base_k = (L>=16)?8:0 ; half h: v=h>>1, j=h&1
// k = ((v>=4)?16:0) + base_k + ((v&3)<<1) + j
__device__ __forceinline__ int a_klocal(int lane, int half) {
    int v = half >> 1, j = half & 1;
    return ((v >= 4) ? 16 : 0) + ((lane >= 16) ? 8 : 0) + ((v & 3) << 1) + j;
}

// ------------------- pack x into WMMA A-layout (f16) -------------------
// layout: [t][kt(0..7)][mtA(0..1)][lane(0..31)][half(0..15)]
__global__ void pack_x_kernel(const float* __restrict__ x, _Float16* __restrict__ xA) {
    int idx = blockIdx.x * blockDim.x + threadIdx.x;
    if (idx >= T_ * KX * 2 * 32 * 16) return;
    int half = idx & 15;
    int lane = (idx >> 4) & 31;
    int mtA  = (idx >> 9) & 1;
    int kt   = (idx >> 10) & 7;
    int t    = idx >> 13;
    int b = mtA * 16 + (lane & 15);
    int f = kt * 32 + a_klocal(lane, half);
    float v = (f < F_) ? x[((size_t)b * T_ + t) * F_ + f] : 0.0f;
    xA[idx] = (_Float16)v;
}

// ------------- pack combined [Wx;Wh] into WMMA B-layout (f16) -------------
// B tile (32x16 KxN): lane&15 = n within tile, (lane>>4, half) -> k via a_klocal
// layout: [lstm][jt(25)][gate(4)][kt(21)][lane][half]; K: 0..255 -> Wx(f<240), 256..671 -> Wh(kh<400)
__global__ void pack_w_kernel(const float* __restrict__ Wx_c, const float* __restrict__ Wh_c,
                              const float* __restrict__ Wx_l, const float* __restrict__ Wh_l,
                              _Float16* __restrict__ pW) {
    int idx = blockIdx.x * blockDim.x + threadIdx.x;
    if (idx >= PW_HALF) return;
    int half = idx & 15;
    int lane = (idx >> 4) & 31;
    int q = idx >> 9;
    int kt = q % KTOT; q /= KTOT;
    int g  = q & 3;    q >>= 2;
    int jt = q % JT;
    int lstm = q / JT;
    const float* Wx = lstm ? Wx_l : Wx_c;
    const float* Wh = lstm ? Wh_l : Wh_c;
    int K = kt * 32 + a_klocal(lane, half);
    int col = g * H_ + (jt * 16 + (lane & 15));
    float v = 0.0f;
    if (K < 256) {
        if (K < F_) v = Wx[(size_t)K * G4H + col];
    } else {
        int kh = K - 256;
        if (kh < H_) v = Wh[(size_t)kh * G4H + col];
    }
    pW[idx] = (_Float16)v;
}

// ------------- pack Wd into WMMA B-layout (f16) -------------
// layout: [lstm][nt(5)][kt(13)][lane][half]
__global__ void pack_wd_kernel(const float* __restrict__ Wd_c, const float* __restrict__ Wd_l,
                               _Float16* __restrict__ pWd) {
    int idx = blockIdx.x * blockDim.x + threadIdx.x;
    if (idx >= PWD_HALF) return;
    int half = idx & 15;
    int lane = (idx >> 4) & 31;
    int q = idx >> 9;
    int kt = q % KH; q /= KH;
    int nt = q % NTD;
    int lstm = q / NTD;
    const float* Wd = lstm ? Wd_l : Wd_c;
    int kh = kt * 32 + a_klocal(lane, half);
    int c  = nt * 16 + (lane & 15);
    float v = (kh < H_ && c < C_) ? Wd[(size_t)kh * C_ + c] : 0.0f;
    pWd[idx] = (_Float16)v;
}

// ------------------------- fused LSTM + dense head -------------------------
// One workgroup (25 waves, wave32) per LSTM. h/c state in registers,
// h tile in LDS in WMMA A-layout, weights streamed from L2 each step
// (rolled k-loop + register double-buffering; no LICM spill-to-scratch).
__global__ __launch_bounds__(800) void lstm_kernel(
    const int*   __restrict__ seq_len,
    const float* __restrict__ b_c,  const float* __restrict__ b_l,
    const float* __restrict__ bd_c, const float* __restrict__ bd_l,
    const _Float16* __restrict__ pW, const _Float16* __restrict__ pWd,
    const _Float16* __restrict__ xA, float* __restrict__ out)
{
    __shared__ _Float16 hA[KH * 2 * 32 * 16];   // 26,624 B
    __shared__ int sl[B_];

    const int lstm = blockIdx.x;
    const int tid  = threadIdx.x;
    const int wave = tid >> 5;
    const int lane = tid & 31;
    const int hi   = (lane >> 4) & 1;

    const float* bias = lstm ? b_l  : b_c;
    const float* bd   = lstm ? bd_l : bd_c;
    const _Float16* pWl  = pW  + (size_t)lstm * (JT * 4 * KTOT * 32 * 16);
    const _Float16* pWdl = pWd + (size_t)lstm * (NTD * KH * 32 * 16);
    float* outl = out + (size_t)lstm * BTC;

    // zero initial h operand, load seq_len
    for (int i = tid; i < KH * 2 * 32 * 16; i += 800) hA[i] = (_Float16)0.0f;
    if (tid < B_) sl[tid] = seq_len[tid];
    __syncthreads();

    // per-lane gate column and bias
    const int j  = wave * 16 + (lane & 15);           // 0..399
    const float bi = bias[0 * H_ + j];
    const float bf = bias[1 * H_ + j];
    const float bg = bias[2 * H_ + j];
    const float bo = bias[3 * H_ + j];

    // precompute hA scatter coordinates for this lane's j column
    const int kt_h   = j >> 5;
    const int klocal = j & 31;
    const int t16    = (klocal >= 16) ? 1 : 0;
    const int k15    = klocal & 15;
    const int base8  = (k15 >= 8) ? 1 : 0;
    const int vv     = 4 * t16 + ((k15 & 7) >> 1);
    const int halfw  = 2 * vv + (k15 & 1);
    const int lanehi = base8 * 16;

    // dense-head assignment (waves 0..9)
    const int mtd  = wave & 1;
    const int ntd  = wave >> 1;
    const int cidx = ntd * 16 + (lane & 15);
    const float bdv = (wave < 10 && cidx < C_) ? bd[cidx] : 0.0f;

    // per-wave weight base (this wave's 4 gate column-tiles)
    const _Float16* wbase = pWl + ((size_t)(wave * 4) * KTOT) * 512 + lane * 16;

    float hs[2][8], cs[2][8];
#pragma unroll
    for (int m = 0; m < 2; ++m)
#pragma unroll
        for (int r = 0; r < 8; ++r) { hs[m][r] = 0.0f; cs[m][r] = 0.0f; }

#pragma clang loop unroll(disable)
    for (int t = 0; t < T_; ++t) {
        // prefetch next timestep's A chunk into near caches
        if (t + 1 < T_)
            __builtin_prefetch(xA + (((size_t)(t + 1) * KX) * 2) * 512 + lane * 16, 0, 3);
#pragma clang loop unroll(disable)
        for (int mt = 0; mt < 2; ++mt) {
            v8f a0 = {}, a1 = {}, a2 = {}, a3 = {};
            const _Float16* aXp = xA + (((size_t)t * KX) * 2 + mt) * 512 + lane * 16;
            // preload kt = 0 operands
            v16h aC = *(const v16h*)aXp;
            v16h w0 = *(const v16h*)(wbase + (size_t)0 * KTOT * 512);
            v16h w1 = *(const v16h*)(wbase + (size_t)1 * KTOT * 512);
            v16h w2 = *(const v16h*)(wbase + (size_t)2 * KTOT * 512);
            v16h w3 = *(const v16h*)(wbase + (size_t)3 * KTOT * 512);
#pragma clang loop unroll(disable)
            for (int kt = 0; kt < KTOT; ++kt) {
                // double-buffer: issue next iteration's loads before consuming
                const int ktn = (kt + 1 < KTOT) ? kt + 1 : kt;   // last iter: redundant reload
                v16h aN;
                if (ktn < KX)
                    aN = *(const v16h*)(aXp + (size_t)ktn * 1024);
                else
                    aN = *(const v16h*)(hA + (((ktn - KX) * 2 + mt) * 32 + lane) * 16);
                const _Float16* wn = wbase + (size_t)ktn * 512;
                v16h w0n = *(const v16h*)(wn + (size_t)0 * KTOT * 512);
                v16h w1n = *(const v16h*)(wn + (size_t)1 * KTOT * 512);
                v16h w2n = *(const v16h*)(wn + (size_t)2 * KTOT * 512);
                v16h w3n = *(const v16h*)(wn + (size_t)3 * KTOT * 512);
                a0 = __builtin_amdgcn_wmma_f32_16x16x32_f16(false, aC, false, w0, (short)0, a0, false, false);
                a1 = __builtin_amdgcn_wmma_f32_16x16x32_f16(false, aC, false, w1, (short)0, a1, false, false);
                a2 = __builtin_amdgcn_wmma_f32_16x16x32_f16(false, aC, false, w2, (short)0, a2, false, false);
                a3 = __builtin_amdgcn_wmma_f32_16x16x32_f16(false, aC, false, w3, (short)0, a3, false, false);
                aC = aN; w0 = w0n; w1 = w1n; w2 = w2n; w3 = w3n;
            }
            __syncthreads();   // all reads of hA[mt] (prev h) complete before overwrite
            // gates: i,f,g,o for rows b = mt*16 + 8*hi + r, column j
#pragma unroll
            for (int r = 0; r < 8; ++r) {
                int b = mt * 16 + 8 * hi + r;
                float zi = a0[r] + bi;
                float zf = a1[r] + bf;
                float zg = a2[r] + bg;
                float zo = a3[r] + bo;
                float cn = sigmf(zf) * cs[mt][r] + sigmf(zi) * tanhf(zg);
                float hn = sigmf(zo) * tanhf(cn);
                if (t < sl[b]) { cs[mt][r] = cn; hs[mt][r] = hn; }
                int lanep = (8 * hi + r) + lanehi;
                hA[((kt_h * 2 + mt) * 32 + lanep) * 16 + halfw] = (_Float16)hs[mt][r];
            }
        }
        __syncthreads();   // full new h visible to dense head
        if (wave < 10) {
            v8f acc = {};
#pragma clang loop unroll(disable)
            for (int kt = 0; kt < KH; ++kt) {
                v16h a = *(const v16h*)(hA + ((kt * 2 + mtd) * 32 + lane) * 16);
                v16h w = *(const v16h*)(pWdl + (((size_t)ntd * KH + kt) * 32 + lane) * 16);
                acc = __builtin_amdgcn_wmma_f32_16x16x32_f16(false, a, false, w, (short)0, acc, false, false);
            }
            if (cidx < C_) {
#pragma unroll
                for (int r = 0; r < 8; ++r) {
                    int b = mtd * 16 + 8 * hi + r;
                    outl[((size_t)b * T_ + t) * C_ + cidx] = acc[r] + bdv;
                }
            }
        }
        __syncthreads();   // dense reads done before next step's gate writes
    }
}

// ------------------------- max |diff| reduction -------------------------
__global__ void init_max_kernel(unsigned int* p) { *p = 0u; }

__global__ void diff_kernel(const float* __restrict__ out, unsigned int* __restrict__ p) {
    __shared__ unsigned int smax;
    if (threadIdx.x == 0) smax = 0u;
    __syncthreads();
    size_t i = (size_t)blockIdx.x * blockDim.x + threadIdx.x;
    float v = 0.0f;
    if (i < (size_t)BTC) v = fabsf(out[i] - out[(size_t)BTC + i]);
    atomicMax(&smax, __float_as_uint(v));   // v >= 0 so bit order == value order
    __syncthreads();
    if (threadIdx.x == 0) atomicMax(p, smax);
}

__global__ void fin_kernel(const unsigned int* __restrict__ p, float* __restrict__ out) {
    out[(size_t)2 * BTC] = __uint_as_float(*p);
}

extern "C" void kernel_launch(void* const* d_in, const int* in_sizes, int n_in,
                              void* d_out, int out_size, void* d_ws, size_t ws_size,
                              hipStream_t stream) {
    (void)in_sizes; (void)n_in; (void)out_size; (void)ws_size;
    const float* x    = (const float*)d_in[0];
    const int*   slen = (const int*)d_in[1];
    const float* Wx_c = (const float*)d_in[2];
    const float* Wh_c = (const float*)d_in[3];
    const float* b_c  = (const float*)d_in[4];
    const float* Wx_l = (const float*)d_in[5];
    const float* Wh_l = (const float*)d_in[6];
    const float* b_l  = (const float*)d_in[7];
    const float* Wd_c = (const float*)d_in[8];
    const float* bd_c = (const float*)d_in[9];
    const float* Wd_l = (const float*)d_in[10];
    const float* bd_l = (const float*)d_in[11];

    char* ws = (char*)d_ws;
    _Float16* pW   = (_Float16*)(ws + OFF_PW);
    _Float16* pWd  = (_Float16*)(ws + OFF_PWD);
    _Float16* xA   = (_Float16*)(ws + OFF_XA);
    unsigned int* maxp = (unsigned int*)(ws + OFF_MAX);
    float* out = (float*)d_out;

    pack_w_kernel<<<(PW_HALF + 255) / 256, 256, 0, stream>>>(Wx_c, Wh_c, Wx_l, Wh_l, pW);
    pack_wd_kernel<<<(PWD_HALF + 255) / 256, 256, 0, stream>>>(Wd_c, Wd_l, pWd);
    pack_x_kernel<<<(XA_HALF + 255) / 256, 256, 0, stream>>>(x, xA);
    init_max_kernel<<<1, 1, 0, stream>>>(maxp);
    lstm_kernel<<<2, 800, 0, stream>>>(slen, b_c, b_l, bd_c, bd_l, pW, pWd, xA, out);
    diff_kernel<<<(BTC + 255) / 256, 256, 0, stream>>>(out, maxp);
    fin_kernel<<<1, 1, 0, stream>>>(maxp, out);
}
